// CrystalGraphConvNet_81423989997746
// MI455X (gfx1250) — matrix operand order repaired and checked by reading, not compiled, not run on hardware
//
#include <hip/hip_runtime.h>
#include <hip/hip_bf16.h>

typedef __attribute__((ext_vector_type(16))) _Float16 v16h;
typedef __attribute__((ext_vector_type(8)))  _Float16 v8h;
typedef __attribute__((ext_vector_type(8)))  float    v8f;

#define NNODES  100000
#define NEDGES  1000000
#define NGRAPHS 2048
#define KP1 192   // padded 169
#define KP2 128
#define KP3 96
#define NTILES (NEDGES/16)
// packed transposed f16 weights per layer: lin[128][192] + m1[96][128] + m2[64][96]
#define WPACK_HALVES (24576 + 12288 + 6144)   // 43008

__device__ __forceinline__ float sp_f(float x) {
  return x > 20.f ? x : log1pf(__expf(x));
}

// A fragment: act row-major [16][stride] f16. Contiguous k order (matches B loads):
// lanes 0-15: K = k0+0..15 ; lanes 16-31: K = k0+16..31
__device__ __forceinline__ v16h load_a(const _Float16* act, int stride, int k0, int r, int hh) {
  const _Float16* p = act + r * stride + k0 + 16 * hh;
  v8h lo = *(const v8h*)p;
  v8h hi = *(const v8h*)(p + 8);
  v16h a;
#pragma unroll
  for (int i = 0; i < 8; i++) { a[i] = lo[i]; a[i + 8] = hi[i]; }
  return a;
}

// ---------------------------------------------------------------- utility
__global__ void zero_f32(float* p, int n) {
  int i = blockIdx.x * blockDim.x + threadIdx.x;
  if (i < n) p[i] = 0.f;
}

// Build packed transposed f16 weights: wpack[l] = linT[128][192] | m1T[96][128] | m2T[64][96]
__global__ void convert_weights(const float* __restrict__ lin_w, const float* __restrict__ m1_w,
                                const float* __restrict__ m2_w, _Float16* __restrict__ wpack) {
  int i = blockIdx.x * blockDim.x + threadIdx.x;
  if (i >= 3 * WPACK_HALVES) return;
  int l = i / WPACK_HALVES, rr = i % WPACK_HALVES;
  float v;
  if (rr < 24576) {
    int n = rr / 192, k = rr % 192;
    v = (k < 169) ? lin_w[l * 169 * 128 + k * 128 + n] : 0.f;
  } else if (rr < 36864) {
    int q = rr - 24576; int n = q / 128, k = q % 128;
    v = m1_w[l * 128 * 96 + k * 96 + n];
  } else {
    int q = rr - 36864; int n = q / 96, k = q % 96;
    v = m2_w[l * 96 * 64 + k * 64 + n];
  }
  wpack[i] = (_Float16)v;
}

// h = x @ emb_w + emb_b  (tiny: plain VALU)
__global__ void embed_k(const float* __restrict__ x, const float* __restrict__ w,
                        const float* __restrict__ b, float* __restrict__ h32,
                        _Float16* __restrict__ h16) {
  __shared__ float xs[92];
  int node = blockIdx.x;
  int f = threadIdx.x;  // 64 threads
  for (int k = f; k < 92; k += 64) xs[k] = x[node * 92 + k];
  __syncthreads();
  float acc = b[f];
  for (int k = 0; k < 92; k++) acc += xs[k] * w[k * 64 + f];
  h32[node * 64 + f] = acc;
  h16[node * 64 + f] = (_Float16)acc;
}

// -------------------------------------------------- fused per-edge MLP (WMMA)
// One wave = one 16-edge tile: z[16x192] -(169->128,+bias)-> a1 -(sp,128->96)-> a2
// -(sp,96->64)-> atomicAdd scatter to agg[row].
// A fragments are cached in registers per tile; all nt accumulators held in registers
// so the WMMA inner loops contain no LDS stores (lets the scheduler pipeline B loads).
__global__ void edge_mlp(const int* __restrict__ ei, const float* __restrict__ eattr,
                         const _Float16* __restrict__ h16,
                         const _Float16* __restrict__ wpackL,
                         const float* __restrict__ linB, const float* __restrict__ m1B,
                         const float* __restrict__ m2B, float* __restrict__ agg) {
  extern __shared__ char smem[];
  _Float16* wLin = (_Float16*)smem;          // [128][192]
  _Float16* wM1  = wLin + 24576;             // [96][128]
  _Float16* wM2  = wM1 + 12288;              // [64][96]
  float* bLin = (float*)(wM2 + 6144);
  float* bM1  = bLin + 128;
  float* bM2  = bM1 + 96;
  _Float16* actB = (_Float16*)(bM2 + 64);    // per-wave: z 3072h | a1 2048h | a2 1536h

  // stage weights + biases for this layer (block-cooperative, once)
  {
    const uint4* src = (const uint4*)wpackL;
    uint4* dst = (uint4*)wLin;
    for (int i = threadIdx.x; i < WPACK_HALVES / 8; i += blockDim.x) dst[i] = src[i];
    for (int i = threadIdx.x; i < 128; i += blockDim.x) bLin[i] = linB[i];
    for (int i = threadIdx.x; i < 96; i += blockDim.x)  bM1[i] = m1B[i];
    for (int i = threadIdx.x; i < 64; i += blockDim.x)  bM2[i] = m2B[i];
  }
  __syncthreads();

  const int lane = threadIdx.x & 31;
  const int wid  = threadIdx.x >> 5;
  const int hh = lane >> 4;       // lane half
  const int r  = lane & 15;
  _Float16* z  = actB + wid * 6656;
  _Float16* a1 = z + 3072;
  _Float16* a2 = a1 + 2048;
  const int nwaves = (blockDim.x >> 5) * gridDim.x;

  for (int tile = blockIdx.x * (blockDim.x >> 5) + wid; tile < NTILES; tile += nwaves) {
    const int e0 = tile * 16;

    // ---- build z tile [16][192] f16: h16[row] | h16[col] | edge_attr | 0-pad
    {
      int e = lane >> 1, part = lane & 1;          // 2 lanes per edge
      int node = ei[part * NEDGES + e0 + e];       // part0=row, part1=col
      const uint4* src = (const uint4*)(h16 + node * 64);
      uint4* dst = (uint4*)(z + e * KP1 + part * 64);
#pragma unroll
      for (int i = 0; i < 8; i++) dst[i] = src[i];
      const float* ap = eattr + (e0 + e) * 41;
      for (int k = part; k < 41; k += 2) z[e * KP1 + 128 + k] = (_Float16)ap[k];
      for (int k = 169 + part; k < KP1; k += 2) z[e * KP1 + k] = (_Float16)0.f;
    }

    // row indices for the 8 edges this lane's D-fragment rows cover
    int rIdx[8];
#pragma unroll
    for (int v = 0; v < 8; v++) rIdx[v] = ei[e0 + v + 8 * hh];

    // ---- GEMM1: [16x192] @ [192x128] + bias -> a1 (f16)
    {
      v16h aF[6];
#pragma unroll
      for (int k = 0; k < 6; k++) aF[k] = load_a(z, KP1, k * 32, r, hh);
      v8f c1[8];
#pragma unroll
      for (int nt = 0; nt < 8; nt++) {
        v8f c = {};
#pragma unroll
        for (int k = 0; k < 6; k++) {
          v16h b = *(const v16h*)(wLin + (nt * 16 + r) * KP1 + k * 32 + 16 * hh);
          c = __builtin_amdgcn_wmma_f32_16x16x32_f16(false, aF[k], false, b, (short)0, c, false, false);
        }
        c1[nt] = c;
      }
#pragma unroll
      for (int nt = 0; nt < 8; nt++) {
        float bias = bLin[nt * 16 + r];
#pragma unroll
        for (int v = 0; v < 8; v++)
          a1[(v + 8 * hh) * KP2 + nt * 16 + r] = (_Float16)(c1[nt][v] + bias);
      }
    }

    // ---- GEMM2: [16x128] @ [128x96] + bias, softplus -> a2 (f16)
    {
      v16h aF[4];
#pragma unroll
      for (int k = 0; k < 4; k++) aF[k] = load_a(a1, KP2, k * 32, r, hh);
      v8f c2[6];
#pragma unroll
      for (int nt = 0; nt < 6; nt++) {
        v8f c = {};
#pragma unroll
        for (int k = 0; k < 4; k++) {
          v16h b = *(const v16h*)(wM1 + (nt * 16 + r) * KP2 + k * 32 + 16 * hh);
          c = __builtin_amdgcn_wmma_f32_16x16x32_f16(false, aF[k], false, b, (short)0, c, false, false);
        }
        c2[nt] = c;
      }
#pragma unroll
      for (int nt = 0; nt < 6; nt++) {
        float bias = bM1[nt * 16 + r];
#pragma unroll
        for (int v = 0; v < 8; v++)
          a2[(v + 8 * hh) * KP3 + nt * 16 + r] = (_Float16)sp_f(c2[nt][v] + bias);
      }
    }

    // ---- GEMM3: [16x96] @ [96x64] + bias, softplus, scatter-add to agg[row]
    {
      v16h aF[3];
#pragma unroll
      for (int k = 0; k < 3; k++) aF[k] = load_a(a2, KP3, k * 32, r, hh);
      v8f c3[4];
#pragma unroll
      for (int nt = 0; nt < 4; nt++) {
        v8f c = {};
#pragma unroll
        for (int k = 0; k < 3; k++) {
          v16h b = *(const v16h*)(wM2 + (nt * 16 + r) * KP3 + k * 32 + 16 * hh);
          c = __builtin_amdgcn_wmma_f32_16x16x32_f16(false, aF[k], false, b, (short)0, c, false, false);
        }
        c3[nt] = c;
      }
#pragma unroll
      for (int nt = 0; nt < 4; nt++) {
        float bias = bM2[nt * 16 + r];
#pragma unroll
        for (int v = 0; v < 8; v++)
          atomicAdd(&agg[rIdx[v] * 64 + nt * 16 + r], sp_f(c3[nt][v] + bias));
      }
    }
  }
}

// -------------------------------------------------- BN stats, BN+residual
__global__ void stats_k(const float* __restrict__ agg, float* __restrict__ stats) {
  int f = threadIdx.x & 63;
  int sub = threadIdx.x >> 6;    // 0..3
  float s = 0.f, s2 = 0.f;
  for (int n = blockIdx.x * 4 + sub; n < NNODES; n += gridDim.x * 4) {
    float t = agg[n * 64 + f];
    s += t; s2 += t * t;
  }
  atomicAdd(&stats[f], s);
  atomicAdd(&stats[64 + f], s2);
}

__global__ void bn_k(const float* __restrict__ agg, const float* __restrict__ stats,
                     const float* __restrict__ g, const float* __restrict__ b,
                     float* __restrict__ h32, _Float16* __restrict__ h16) {
  int i = blockIdx.x * blockDim.x + threadIdx.x;
  if (i >= NNODES * 64) return;
  int f = i & 63;
  float mu  = stats[f] * (1.f / NNODES);
  float var = stats[64 + f] * (1.f / NNODES) - mu * mu;
  float bn = (agg[i] - mu) * rsqrtf(var + 1e-5f) * g[f] + b[f];
  float hn = sp_f(bn + h32[i]);
  h32[i] = hn;
  h16[i] = (_Float16)hn;
}

// -------------------------------------------------- graph pooling + head
__global__ void pool_k(const float* __restrict__ h32, const int* __restrict__ batch,
                       float* __restrict__ gsum, int* __restrict__ gmax,
                       float* __restrict__ cnt) {
  int i = blockIdx.x * blockDim.x + threadIdx.x;
  if (i >= NNODES * 64) return;
  int n = i >> 6, f = i & 63;
  int bg = batch[n];
  float v = h32[i];
  atomicAdd(&gsum[bg * 64 + f], v);
  atomicMax(&gmax[bg * 64 + f], __float_as_int(v));  // h > 0 (softplus) -> int order OK
  if (f == 0) atomicAdd(&cnt[bg], 1.f);
}

__global__ void head_k(const float* __restrict__ gsum, const int* __restrict__ gmax,
                       const float* __restrict__ cnt,
                       const float* __restrict__ fcw, const float* __restrict__ fcb,
                       const float* __restrict__ ow, const float* __restrict__ ob,
                       float* __restrict__ out) {
  __shared__ float s[128];
  __shared__ float cf[128];
  int g = blockIdx.x, t = threadIdx.x;  // 128 threads
  float crys;
  if (t < 64) crys = gsum[g * 64 + t] / fmaxf(cnt[g], 1.f);
  else        crys = __int_as_float(gmax[g * 64 + t - 64]);
  out[NGRAPHS + g * 128 + t] = crys;          // crys_fea_inter output
  s[t] = sp_f(crys);
  __syncthreads();
  float acc = fcb[t];
  for (int k = 0; k < 128; k++) acc += s[k] * fcw[k * 128 + t];
  cf[t] = sp_f(acc);
  __syncthreads();
  if (t == 0) {
    float o = ob[0];
    for (int k = 0; k < 128; k++) o += cf[k] * ow[k];
    out[g] = o;                                // out output
  }
}

extern "C" void kernel_launch(void* const* d_in, const int* in_sizes, int n_in,
                              void* d_out, int out_size, void* d_ws, size_t ws_size,
                              hipStream_t stream) {
  const float* x      = (const float*)d_in[0];
  const int*   ei     = (const int*)d_in[1];
  const float* eattr  = (const float*)d_in[2];
  const int*   batch  = (const int*)d_in[3];
  const float* emb_w  = (const float*)d_in[4];
  const float* emb_b  = (const float*)d_in[5];
  const float* lin_w  = (const float*)d_in[6];
  const float* lin_b  = (const float*)d_in[7];
  const float* m1_w   = (const float*)d_in[8];
  const float* m1_b   = (const float*)d_in[9];
  const float* m2_w   = (const float*)d_in[10];
  const float* m2_b   = (const float*)d_in[11];
  const float* bn_g   = (const float*)d_in[12];
  const float* bn_b   = (const float*)d_in[13];
  const float* fc_w   = (const float*)d_in[14];
  const float* fc_b   = (const float*)d_in[15];
  const float* out_w  = (const float*)d_in[16];
  const float* out_b  = (const float*)d_in[17];

  char* ws = (char*)d_ws;
  float*    h32   = (float*)ws;     ws += (size_t)NNODES * 64 * 4;
  _Float16* h16   = (_Float16*)ws;  ws += (size_t)NNODES * 64 * 2;
  float*    agg   = (float*)ws;     ws += (size_t)NNODES * 64 * 4;
  float*    stats = (float*)ws;     ws += 512;
  float*    gsum  = (float*)ws;     ws += (size_t)NGRAPHS * 64 * 4;
  int*      gmax  = (int*)ws;       ws += (size_t)NGRAPHS * 64 * 4;
  float*    cnt   = (float*)ws;     ws += (size_t)NGRAPHS * 4;
  _Float16* wpack = (_Float16*)ws;  ws += (size_t)3 * WPACK_HALVES * 2;

  convert_weights<<<(3 * WPACK_HALVES + 255) / 256, 256, 0, stream>>>(lin_w, m1_w, m2_w, wpack);
  embed_k<<<NNODES, 64, 0, stream>>>(x, emb_w, emb_b, h32, h16);

  const int SMEM = WPACK_HALVES * 2 + (128 + 96 + 64) * 4 + 4 * 6656 * 2;  // 140416 B
  for (int l = 0; l < 3; l++) {
    zero_f32<<<(NNODES * 64 + 255) / 256, 256, 0, stream>>>(agg, NNODES * 64);
    zero_f32<<<1, 128, 0, stream>>>(stats, 128);
    edge_mlp<<<1024, 128, SMEM, stream>>>(ei, eattr, h16, wpack + (size_t)l * WPACK_HALVES,
                                          lin_b + l * 128, m1_b + l * 96, m2_b + l * 64, agg);
    stats_k<<<512, 256, 0, stream>>>(agg, stats);
    bn_k<<<(NNODES * 64 + 255) / 256, 256, 0, stream>>>(agg, stats, bn_g + l * 64,
                                                        bn_b + l * 64, h32, h16);
  }

  // gsum | gmax | cnt are contiguous: zero all at once
  zero_f32<<<(NGRAPHS * 129 + 255) / 256, 256, 0, stream>>>(gsum, NGRAPHS * 129);
  pool_k<<<(NNODES * 64 + 255) / 256, 256, 0, stream>>>(h32, batch, gsum, gmax, cnt);
  head_k<<<NGRAPHS, 128, 0, stream>>>(gsum, gmax, cnt, fc_w, fc_b, out_w, out_b, (float*)d_out);
}
